// AdversarialGAT_61864708932049
// MI455X (gfx1250) — matrix operand order repaired and checked by reading, not compiled, not run on hardware
//
#include <hip/hip_runtime.h>

typedef __attribute__((ext_vector_type(16))) _Float16 v16h;
typedef __attribute__((ext_vector_type(8)))  float    v8f;

namespace {
constexpr int NNODE  = 50000;
constexpr int NEDGE  = 800000;
constexpr int NGRAPH = 256;
constexpr int DEMB   = 32;
constexpr int VOCAB  = 120;
constexpr int DH     = 100;
constexpr int DFC    = 32;
constexpr int CAP    = 25;
constexpr int NTYPE  = 119;   // types 1..119
constexpr int MINT   = 5;
}

// K index inside a 16x32 f16 A-fragment per CDNA5 ISA 7.12.2:
// lanes 0-15 cover K {0..7, 16..23}, lanes 16-31 cover K {8..15, 24..31}
__device__ __forceinline__ int a_kmap(int e, int lane) {
  int j = e >> 1;  // VGPR 0..7
  return ((j >= 4) ? 16 : 0) + ((lane >> 4) << 3) + ((j & 3) << 1) + (e & 1);
}

__global__ void k_zero(float* p, int n) {
  int stride = gridDim.x * blockDim.x;
  for (int i = blockIdx.x * blockDim.x + threadIdx.x; i < n; i += stride) p[i] = 0.f;
}

__global__ void k_deg(const int* __restrict__ ei, float* outdeg, float* indeg) {
  int stride = gridDim.x * blockDim.x;
  for (int e = blockIdx.x * blockDim.x + threadIdx.x; e < NEDGE; e += stride) {
    atomicAdd(&outdeg[ei[e]], 1.f);
    atomicAdd(&indeg[ei[NEDGE + e]], 1.f);
  }
}

// hw[i,:] = (embed[feat[i]] * mask[i]) @ gcn_w * outdeg[i]^-1/2   (K=32 == one WMMA)
// A and B fragments staged in LDS pre-swizzled: one contiguous v16h per lane.
__global__ void k_gemm_embed(const int* __restrict__ feat,
                             const float* __restrict__ embed,
                             const float* __restrict__ mask,
                             const float* __restrict__ w,      // [32][100]
                             const float* __restrict__ outdeg,
                             float* __restrict__ hw, int M) {
  __shared__ v16h BsSw[7][32];        // [ct][lane] -> contiguous fragment
  __shared__ v16h EmbSw[2][VOCAB];    // [lane>>4][type] -> contiguous A fragment
  int tid = threadIdx.x;
  {
    _Float16* bp = (_Float16*)BsSw;
    for (int idx = tid; idx < 7 * 32 * 16; idx += blockDim.x) {
      int ct = idx >> 9, rem = idx & 511;
      int lane = rem >> 4, e = rem & 15;
      int K = ((e >> 3) << 4) + ((lane >> 4) << 3) + (e & 7);
      int c = ct * 16 + (lane & 15);
      bp[idx] = (c < DH) ? (_Float16)w[K * DH + c] : (_Float16)0.f;
    }
    _Float16* ep = (_Float16*)EmbSw;
    for (int idx = tid; idx < 2 * VOCAB * 16; idx += blockDim.x) {
      int g = idx / (VOCAB * 16), rem = idx - g * (VOCAB * 16);
      int f = rem >> 4, e = rem & 15;
      ep[idx] = (_Float16)embed[f * DEMB + a_kmap(e, g << 4)];
    }
  }
  __syncthreads();
  int tile = blockIdx.x * 4 + (tid >> 5);          // wave-uniform
  if (tile >= M / 16) return;
  int lane = tid & 31;
  int rowBase = tile * 16;
  int row = rowBase + (lane & 15);
  v16h a = EmbSw[lane >> 4][feat[row]];
  if (mask) a = a * (_Float16)mask[row];           // packed f16 muls
  float rsv[8];
#pragma unroll
  for (int r = 0; r < 8; ++r) {
    int orow = rowBase + r + ((lane >> 4) << 3);
    rsv[r] = rsqrtf(fmaxf(outdeg[orow], 1.f));
  }
  int cl = lane & 15;
  for (int ct = 0; ct < 7; ++ct) {
    v16h b = BsSw[ct][lane];                       // 2x ds_load_b128
    v8f c = {};
    c = __builtin_amdgcn_wmma_f32_16x16x32_f16(false, a, false, b, (short)0, c, false, false);
#pragma unroll
    for (int r = 0; r < 8; ++r) {
      int orow = rowBase + r + ((lane >> 4) << 3);
      int col = ct * 16 + cl;
      if (col < DH) hw[(size_t)orow * DH + col] = c[r] * rsv[r];
    }
  }
}

// agg[dst,:] += hw[src,:] over edges; one wave per edge, lanes stride columns
__global__ void k_agg(const int* __restrict__ ei, const float* __restrict__ hw,
                      float* __restrict__ agg) {
  int wave = (blockIdx.x * blockDim.x + threadIdx.x) >> 5;
  int lane = threadIdx.x & 31;
  int nwave = (gridDim.x * blockDim.x) >> 5;
  for (int e = wave; e < NEDGE; e += nwave) {
    int s = ei[e], d = ei[NEDGE + e];
    const float* hs = hw + (size_t)s * DH;
    float* ad = agg + (size_t)d * DH;
    for (int c = lane; c < DH; c += 32) atomicAdd(&ad[c], hs[c]);
  }
}

// h = relu(agg * indeg^-1/2 + b), in place; pooled[g] = segment_max(h)  (h>=0 -> uint max trick)
__global__ void k_norm_pool(float* __restrict__ h, const float* __restrict__ indeg,
                            const float* __restrict__ b, const int* __restrict__ gid,
                            float* __restrict__ pooled) {
  int stride = gridDim.x * blockDim.x;
  for (int idx = blockIdx.x * blockDim.x + threadIdx.x; idx < NNODE * DH; idx += stride) {
    int i = idx / DH, c = idx - i * DH;
    float v = fmaxf(h[idx] * rsqrtf(fmaxf(indeg[i], 1.f)) + b[c], 0.f);
    h[idx] = v;
    atomicMax((unsigned int*)&pooled[(size_t)gid[i] * DH + c], __float_as_uint(v));
  }
}

// out[M][32] = A[M][100] @ w[100][32] + bias   (K padded to 128 -> 4 accumulating WMMAs)
__global__ void k_gemm_fc(const float* __restrict__ A,
                          const float* __restrict__ w, const float* __restrict__ bias,
                          float* __restrict__ out, int M) {
  __shared__ v16h BsSw[2][4][32];     // [ct][kc][lane] -> contiguous fragment
  int tid = threadIdx.x;
  {
    _Float16* bp = (_Float16*)BsSw;
    for (int idx = tid; idx < 2 * 4 * 32 * 16; idx += blockDim.x) {
      int ct = idx >> 11, rem = idx & 2047;
      int kc = rem >> 9, rem2 = rem & 511;
      int lane = rem2 >> 4, e = rem2 & 15;
      int K = kc * 32 + ((e >> 3) << 4) + ((lane >> 4) << 3) + (e & 7);
      int c = ct * 16 + (lane & 15);
      bp[idx] = (K < DH) ? (_Float16)w[K * DFC + c] : (_Float16)0.f;
    }
  }
  __syncthreads();
  int nTiles = (M + 15) >> 4;
  int tile = blockIdx.x * 4 + (tid >> 5);          // wave-uniform
  if (tile >= nTiles) return;
  int lane = tid & 31;
  int rowBase = tile * 16;
  int row = rowBase + (lane & 15);
  int rowc = min(row, M - 1);
  v16h afrag[4];
#pragma unroll
  for (int kc = 0; kc < 4; ++kc) {
#pragma unroll
    for (int e = 0; e < 16; ++e) {
      int K = kc * 32 + a_kmap(e, lane);           // two contiguous 8-float runs
      afrag[kc][e] = (K < DH && row < M) ? (_Float16)A[(size_t)rowc * DH + K]
                                         : (_Float16)0.f;
    }
  }
  int cl = lane & 15;
  for (int ct = 0; ct < 2; ++ct) {
    v8f acc = {};
#pragma unroll
    for (int kc = 0; kc < 4; ++kc) {
      v16h b = BsSw[ct][kc][lane];                 // 2x ds_load_b128
      acc = __builtin_amdgcn_wmma_f32_16x16x32_f16(false, afrag[kc], false, b,
                                                   (short)0, acc, false, false);
    }
#pragma unroll
    for (int r = 0; r < 8; ++r) {
      int orow = rowBase + r + ((lane >> 4) << 3);
      int col = ct * 16 + cl;
      if (orow < M) out[(size_t)orow * DFC + col] = acc[r] + bias[col];
    }
  }
}

__global__ void k_defect(const float* __restrict__ xb, const float* __restrict__ dw,
                         const float* __restrict__ db, float* __restrict__ out) {
  int r = blockIdx.x * blockDim.x + threadIdx.x;
  if (r >= NGRAPH) return;
  float s = db[0];
#pragma unroll
  for (int c = 0; c < DFC; ++c) s += xb[r * DFC + c] * dw[c];
  out[r] = 1.f / (1.f + expf(-s));
}

// first CAP indices per type, in index order (== jax top_k tie-break), wave32 ballot scan
__global__ void k_collect(const int* __restrict__ fsrc, const int* __restrict__ ftar,
                          int* __restrict__ idxbuf, int* __restrict__ cnts) {
  int t = blockIdx.x;          // type t+1
  int side = blockIdx.y;       // 0=src, 1=tar
  const int* f = side ? ftar : fsrc;
  int* myidx = idxbuf + (side * NTYPE + t) * CAP;
  int lane = threadIdx.x;
  int type = t + 1;
  int count = 0;
  for (int base = 0; base < NNODE; base += 32) {
    int i = base + lane;
    bool pred = (i < NNODE) && (f[i] == type);
    unsigned long long m = __ballot(pred);
    int prefix = __popcll(m & ((1ull << lane) - 1ull));
    if (pred && (count + prefix) < CAP) myidx[count + prefix] = i;
    count += __popcll(m);
  }
  if (lane == 0) cnts[side * NTYPE + t] = count;
}

__global__ void k_mmd(const float* __restrict__ nfs, const float* __restrict__ nft,
                      const int* __restrict__ sidx, const int* __restrict__ tidx,
                      const int* __restrict__ scnts, const int* __restrict__ tcnts,
                      float* __restrict__ accum) {
  __shared__ float Xs[2 * CAP][DFC + 1];
  __shared__ float D2[2 * CAP][2 * CAP];
  __shared__ float red[256];
  int t = blockIdx.x;
  int tid = threadIdx.x;
  int scnt = scnts[t], tcnt = tcnts[t];
  int m = min(scnt, CAP), n = min(tcnt, CAP);
  for (int idx = tid; idx < 2 * CAP * DFC; idx += blockDim.x) {
    int i = idx / DFC, k = idx - (idx / DFC) * DFC;
    float v = 0.f;
    if (i < CAP) { if (i < m) v = nfs[(size_t)sidx[t * CAP + i] * DFC + k]; }
    else { int j = i - CAP; if (j < n) v = nft[(size_t)tidx[t * CAP + j] * DFC + k]; }
    Xs[i][k] = v;
  }
  __syncthreads();
  float lsum = 0.f;
  for (int p = tid; p < 4 * CAP * CAP; p += blockDim.x) {
    int i = p / (2 * CAP), j = p - i * (2 * CAP);
    float d2 = 0.f;
    for (int k = 0; k < DFC; ++k) { float df = Xs[i][k] - Xs[j][k]; d2 += df * df; }
    D2[i][j] = d2;
    bool vi = (i < CAP) ? (i < m) : ((i - CAP) < n);
    bool vj = (j < CAP) ? (j < m) : ((j - CAP) < n);
    if (vi && vj) lsum += d2;
  }
  red[tid] = lsum; __syncthreads();
  for (int s = 128; s > 0; s >>= 1) { if (tid < s) red[tid] += red[tid + s]; __syncthreads(); }
  float ntot = (float)(m + n);
  float bw = red[0] / fmaxf(ntot * ntot - ntot, 1.f);
  bw *= 0.25f;                               // / KER_MUL^(KER_NUM//2)
  __syncthreads();
  float xx = 0.f, yy = 0.f, xy = 0.f;
  for (int p = tid; p < 4 * CAP * CAP; p += blockDim.x) {
    int i = p / (2 * CAP), j = p - i * (2 * CAP);
    bool vi = (i < CAP) ? (i < m) : ((i - CAP) < n);
    bool vj = (j < CAP) ? (j < m) : ((j - CAP) < n);
    if (!(vi && vj)) continue;
    float d2 = D2[i][j];
    float kv = 0.f, bwk = bw;
#pragma unroll
    for (int q = 0; q < 5; ++q) { kv += expf(-d2 / fmaxf(bwk, 1e-8f)); bwk *= 2.f; }
    if (i < CAP && j < CAP) xx += kv;
    else if (i >= CAP && j >= CAP) yy += kv;
    else if (i < CAP) xy += kv;
  }
  red[tid] = xx; __syncthreads();
  for (int s = 128; s > 0; s >>= 1) { if (tid < s) red[tid] += red[tid + s]; __syncthreads(); }
  float XXs = red[0]; __syncthreads();
  red[tid] = yy; __syncthreads();
  for (int s = 128; s > 0; s >>= 1) { if (tid < s) red[tid] += red[tid + s]; __syncthreads(); }
  float YYs = red[0]; __syncthreads();
  red[tid] = xy; __syncthreads();
  for (int s = 128; s > 0; s >>= 1) { if (tid < s) red[tid] += red[tid + s]; __syncthreads(); }
  float XYs = red[0];
  if (tid == 0) {
    float fm = (float)m, fn = (float)n;
    float loss = XXs / fmaxf(fm * fm, 1.f) + YYs / fmaxf(fn * fn, 1.f)
               - 2.f * XYs / fmaxf(fm * fn, 1.f);
    if (scnt >= MINT && tcnt >= MINT) { atomicAdd(&accum[0], loss); atomicAdd(&accum[1], 1.f); }
  }
}

__global__ void k_final(const float* accum, float* out) {
  if (threadIdx.x == 0 && blockIdx.x == 0)
    out[0] = (accum[1] > 0.f) ? accum[0] / fmaxf(accum[1], 1.f) : 0.f;
}

extern "C" void kernel_launch(void* const* d_in, const int* in_sizes, int n_in,
                              void* d_out, int out_size, void* d_ws, size_t ws_size,
                              hipStream_t stream) {
  const int*   feat     = (const int*)d_in[0];
  const int*   feat_src = (const int*)d_in[1];
  const int*   feat_tar = (const int*)d_in[2];
  const int*   ei       = (const int*)d_in[3];
  const int*   ei_src   = (const int*)d_in[4];
  const int*   ei_tar   = (const int*)d_in[5];
  const int*   gid      = (const int*)d_in[6];
  const float* mask     = (const float*)d_in[7];
  const float* embed    = (const float*)d_in[8];
  const float* gcn_w    = (const float*)d_in[9];
  const float* gcn_b    = (const float*)d_in[10];
  const float* fc_w     = (const float*)d_in[11];
  const float* fc_b     = (const float*)d_in[12];
  const float* def_w    = (const float*)d_in[13];
  const float* def_b    = (const float*)d_in[14];
  float* out = (float*)d_out;

  // workspace layout (floats) — total ~73.5 MB, fits 192 MB L2 for the scatter phase
  float* ws     = (float*)d_ws;
  float* hw     = ws;                 // [N][100] pre-aggregation features
  float* bufA   = ws + 5000000;       // h_src (agg in place)
  float* bufB   = ws + 10000000;      // batch transient agg, then h_tar
  float* outdeg = ws + 15000000;      // [N]
  float* indeg  = ws + 15050000;      // [N]
  float* pooled = ws + 15100000;      // 3*[256][100]
  float* nfs    = ws + 15176800;      // [N][32] src node feats
  float* nft    = ws + 16776800;      // [N][32] tgt node feats
  float* accum  = ws + 18376800;      // {loss_sum, inc_count}
  int*   idxbuf = (int*)(ws + 18376802);        // 2*119*25 indices
  int*   cnts   = idxbuf + 2 * NTYPE * CAP;     // 2*119 counts

  const int tileBlocks = (NNODE / 16 + 3) / 4;  // 782

  auto pass = [&](const int* f, const int* e, const float* mk, float* agg, float* pl) {
    k_zero<<<200, 256, 0, stream>>>(outdeg, NNODE);
    k_zero<<<200, 256, 0, stream>>>(indeg, NNODE);
    k_deg<<<1024, 256, 0, stream>>>(e, outdeg, indeg);
    k_gemm_embed<<<tileBlocks, 128, 0, stream>>>(f, embed, mk, gcn_w, outdeg, hw, NNODE);
    k_zero<<<2048, 256, 0, stream>>>(agg, NNODE * DH);
    k_agg<<<4096, 256, 0, stream>>>(e, hw, agg);
    k_norm_pool<<<2048, 256, 0, stream>>>(agg, indeg, gcn_b, gid, pl);
  };

  k_zero<<<75, 256, 0, stream>>>(pooled, 3 * NGRAPH * DH);
  pass(feat,     ei,     mask,    bufB, pooled);                   // batched graph
  pass(feat_src, ei_src, nullptr, bufA, pooled + NGRAPH * DH);     // source
  pass(feat_tar, ei_tar, nullptr, bufB, pooled + 2 * NGRAPH * DH); // target

  // out layout: [x_batch_mmd 8192][defect 256][x_src_mmd 8192][x_tar_mmd 8192][scalar]
  k_gemm_fc<<<4, 128, 0, stream>>>(pooled,                 fc_w, fc_b, out + 0,     NGRAPH);
  k_gemm_fc<<<4, 128, 0, stream>>>(pooled + NGRAPH * DH,   fc_w, fc_b, out + 8448,  NGRAPH);
  k_gemm_fc<<<4, 128, 0, stream>>>(pooled + 2*NGRAPH*DH,   fc_w, fc_b, out + 16640, NGRAPH);
  k_gemm_fc<<<tileBlocks, 128, 0, stream>>>(bufA, fc_w, fc_b, nfs, NNODE);
  k_gemm_fc<<<tileBlocks, 128, 0, stream>>>(bufB, fc_w, fc_b, nft, NNODE);
  k_defect<<<1, 256, 0, stream>>>(out, def_w, def_b, out + 8192);

  k_collect<<<dim3(NTYPE, 2), 32, 0, stream>>>(feat_src, feat_tar, idxbuf, cnts);
  k_zero<<<1, 64, 0, stream>>>(accum, 2);
  k_mmd<<<NTYPE, 256, 0, stream>>>(nfs, nft, idxbuf, idxbuf + NTYPE * CAP,
                                   cnts, cnts + NTYPE, accum);
  k_final<<<1, 1, 0, stream>>>(accum, out + 24832);

  (void)in_sizes; (void)n_in; (void)out_size; (void)ws_size;
}
